// Attention_17841294148316
// MI455X (gfx1250) — matrix operand order repaired and checked by reading, not compiled, not run on hardware
//
#include <hip/hip_runtime.h>
#include <hip/hip_bf16.h>
#include <stdint.h>

typedef _Float16 f16;
typedef _Float16 v16h __attribute__((ext_vector_type(16)));
typedef _Float16 v8h  __attribute__((ext_vector_type(8)));
typedef float    v8f  __attribute__((ext_vector_type(8)));
typedef unsigned int u32x4 __attribute__((ext_vector_type(4)));
typedef int          i32x8 __attribute__((ext_vector_type(8)));
typedef int          i32x4 __attribute__((ext_vector_type(4)));

#define N_GENES 16384
#define Z_DIM   1024
#define D_K     128
#define N_CELLS 4096

// ---------------------------------------------------------------------------
// Deterministic per-element Gumbel(0,1) noise from an integer hash.
// (JAX's threefry stream cannot be bit-matched; statistically equivalent.)
// ---------------------------------------------------------------------------
__device__ __forceinline__ float gumbel_noise(uint32_t idx) {
    uint32_t x = idx + 0x9E3779B9u;
    x ^= x >> 16; x *= 0x7FEB352Du;
    x ^= x >> 15; x *= 0x846CA68Bu;
    x ^= x >> 16;
    float u = (float)(x >> 8) * (1.0f / 16777216.0f) + 5.9604645e-8f;
    return -__logf(-__logf(u));
}

// ---------------------------------------------------------------------------
// Kernel A1: K (z_dim x d_k) f32 -> f16, same layout.
// ---------------------------------------------------------------------------
__global__ __launch_bounds__(256) void convert_k_kernel(
        const float* __restrict__ K, f16* __restrict__ Kh) {
    int i = blockIdx.x * 256 + threadIdx.x;       // grid covers Z_DIM*D_K exactly
    Kh[i] = (f16)K[i];
}

// ---------------------------------------------------------------------------
// Kernel A2: V (z_dim x n_cells) f32 -> Vt (n_cells x z_dim) f16 via LDS tile
// transpose, so recon's B-fragments are contiguous 32B loads per lane.
// ---------------------------------------------------------------------------
__global__ __launch_bounds__(256) void transpose_v_kernel(
        const float* __restrict__ V, f16* __restrict__ Vt) {
    __shared__ float t[32][33];
    const int lx = threadIdx.x & 31;
    const int ly = threadIdx.x >> 5;              // 0..7
    const int c0 = blockIdx.x * 32;               // cell base
    const int z0 = blockIdx.y * 32;               // z base
#pragma unroll
    for (int i = 0; i < 4; ++i) {
        int z = z0 + ly + i * 8;
        t[ly + i * 8][lx] = V[(size_t)z * N_CELLS + c0 + lx];
    }
    __syncthreads();
#pragma unroll
    for (int i = 0; i < 4; ++i) {
        int c = c0 + ly + i * 8;
        Vt[(size_t)c * Z_DIM + z0 + lx] = (f16)t[lx][ly + i * 8];
    }
}

// ---------------------------------------------------------------------------
// Kernel B: scores = QK^T/sqrt(dk) + gumbel, softmax over z_dim.
// Writes p_attn f32 (output) and p_attn f16 (workspace, feeds recon GEMM).
// 1 block = 16 rows. 256 threads = 8 waves; wave w owns n-tiles [8w, 8w+8).
// ---------------------------------------------------------------------------
__global__ __launch_bounds__(256) void scores_softmax_kernel(
        const float* __restrict__ Q, const f16* __restrict__ Kh,
        float* __restrict__ Pout, f16* __restrict__ Ph) {
    __shared__ __align__(16) f16 qs[16 * D_K];    // 4 KB
    __shared__ float ss[16][Z_DIM];               // 64 KB scores/probs
    __shared__ float red[16][16];                 // 1 KB reductions
    __shared__ float rowstat[16];

    const int tid = threadIdx.x;
    const int m0  = blockIdx.x * 16;

    // Load + convert Q tile (16x128 f32 -> f16 LDS), 8 elems/thread, coalesced.
    {
        int e   = tid * 8;
        int row = e >> 7, col = e & 127;
        const float4* src = (const float4*)(Q + (size_t)(m0 + row) * D_K + col);
        float4 x = src[0], y = src[1];
        v8h h;
        h[0]=(f16)x.x; h[1]=(f16)x.y; h[2]=(f16)x.z; h[3]=(f16)x.w;
        h[4]=(f16)y.x; h[5]=(f16)y.y; h[6]=(f16)y.z; h[7]=(f16)y.w;
        *(v8h*)&qs[row * D_K + col] = h;
    }
    __syncthreads();

    const int w = tid >> 5, lane = tid & 31;
    const int lane_lo = lane & 15, kh2 = lane >> 4;
    const float scale = 0.08838834764831845f;     // 1/sqrt(128)

    for (int t = 0; t < 8; ++t) {
        const int n0 = (w * 8 + t) * 16;
        v8f c = {};
#pragma unroll
        for (int kc = 0; kc < 4; ++kc) {
            const int kk = kc * 32;
            // A fragment: lane<16 holds K {0..15}; lane>=16 {16..31}
            v16h a = *(const v16h*)&qs[lane_lo * D_K + kk + kh2 * 16];
            // B fragment: lane holds one K-row (column of K^T), 16 contiguous K.
            v16h b = *(const v16h*)(Kh + (size_t)(n0 + lane_lo) * D_K + kk + kh2 * 16);
            c = __builtin_amdgcn_wmma_f32_16x16x32_f16(
                    false, a, false, b, (short)0, c, false, false);
        }
        // C layout: vgpr r, lane<16 -> (M=r, N=lane), lane>=16 -> (M=r+8).
#pragma unroll
        for (int r = 0; r < 8; ++r) {
            int row = r + kh2 * 8;
            int col = n0 + lane_lo;
            uint32_t gi = (uint32_t)(m0 + row) * Z_DIM + (uint32_t)col;
            ss[row][col] = c[r] * scale + gumbel_noise(gi);   // tau == 1.0
        }
    }
    __syncthreads();

    // Row softmax: 16 threads per row.
    const int row = tid >> 4, sub = tid & 15;
    float m = -3.4e38f;
    for (int j = sub; j < Z_DIM; j += 16) m = fmaxf(m, ss[row][j]);
    red[row][sub] = m;
    __syncthreads();
    if (sub == 0) {
        float mm = red[row][0];
#pragma unroll
        for (int k = 1; k < 16; ++k) mm = fmaxf(mm, red[row][k]);
        rowstat[row] = mm;
    }
    __syncthreads();
    const float rm = rowstat[row];
    float s = 0.0f;
    for (int j = sub; j < Z_DIM; j += 16) {
        float e = __expf(ss[row][j] - rm);
        ss[row][j] = e;
        s += e;
    }
    __syncthreads();
    red[row][sub] = s;
    __syncthreads();
    if (sub == 0) {
        float tt = 0.0f;
#pragma unroll
        for (int k = 0; k < 16; ++k) tt += red[row][k];
        rowstat[row] = tt;
    }
    __syncthreads();
    const float inv = 1.0f / rowstat[row];
    for (int j = sub; j < Z_DIM; j += 16) {
        float p = ss[row][j] * inv;
        size_t o = (size_t)(m0 + row) * Z_DIM + j;
        Pout[o] = p;
        Ph[o]   = (f16)p;
    }
}

// ---------------------------------------------------------------------------
// Kernel C: recon = p_attn @ V.  Block tile 64(M) x 128(N), 8 waves = 4M x 2N.
// A chunks (64x32 f16) are DMA'd into LDS by the Tensor Data Mover
// (tensor_load_to_lds), double-buffered with s_wait_tensorcnt; B fragments
// read directly from L2-resident Vt.  32 K-steps of WMMA per accumulator.
// ---------------------------------------------------------------------------
__global__ __launch_bounds__(256) void recon_kernel(
        const f16* __restrict__ Ph, const f16* __restrict__ Vt,
        float* __restrict__ out) {
    __shared__ __align__(16) f16 abuf[2][64 * 32];    // 2 x 4 KB A chunks

    const int tid  = threadIdx.x;
    const int w    = tid >> 5, lane = tid & 31;
    const int wm   = w & 3,  wn   = w >> 2;
    const int lane_lo = lane & 15, kh2 = lane >> 4;
    const int m0 = blockIdx.y * 64;
    const int n0 = blockIdx.x * 128;
    const int arow = wm * 16 + lane_lo;

    v8f acc[4] = {v8f{}, v8f{}, v8f{}, v8f{}};

    const f16* bbase = Vt + (size_t)(n0 + wn * 64 + lane_lo) * Z_DIM + kh2 * 16;

    // ---- TDM descriptor pieces (workgroup-uniform -> SGPRs) ----
    const uint64_t pbase = (uint64_t)(uintptr_t)Ph + (uint64_t)m0 * Z_DIM * sizeof(f16);
    const uint32_t lds0  = (uint32_t)(uintptr_t)&abuf[0][0];
    const uint32_t lds1  = (uint32_t)(uintptr_t)&abuf[1][0];

    auto issue_tdm = [&](int kk, uint32_t lds_addr) {
        const uint64_t ga = pbase + (uint64_t)kk * sizeof(f16);
        u32x4 g0;
        g0[0] = 1u;                                          // count=1, user mode
        g0[1] = lds_addr;                                    // lds_addr [63:32]
        g0[2] = (uint32_t)ga;                                // global_addr lo
        g0[3] = ((uint32_t)(ga >> 32) & 0x01FFFFFFu)         // global_addr hi
              | 0x80000000u;                                 // type=2 ("image")
        i32x8 g1;
        g1[0] = 0x00010000;      // wg_mask=0, data_size=1 (2B), no barrier/iter/pad
        g1[1] = (int)((uint32_t)(Z_DIM & 0xFFFF) << 16);     // tensor_dim0 lo16
        g1[2] = (int)((uint32_t)(N_GENES & 0xFFFF) << 16);   // dim0 hi=0 | dim1 lo16
        g1[3] = (int)(32u << 16);                            // dim1 hi=0 | tile_dim0=32
        g1[4] = 64;                                          // tile_dim1=64, tile_dim2=0
        g1[5] = Z_DIM;                                       // tensor_dim0_stride=1024
        g1[6] = 0;                                           // stride hi | dim1_stride lo
        g1[7] = 0;
        i32x4 gz4 = {0, 0, 0, 0};                            // groups 2/3: 2D -> zero
        i32x8 gz8 = {0, 0, 0, 0, 0, 0, 0, 0};
        __builtin_amdgcn_tensor_load_to_lds(g0, g1, gz4, gz4, gz8, 0);
    };

    if (w == 0) issue_tdm(0, lds0);                          // prologue: chunk 0

    for (int kc = 0; kc < 32; ++kc) {
        const int kk  = kc * 32;
        const int cur = kc & 1;
        if (w == 0) __builtin_amdgcn_s_wait_tensorcnt(0);    // chunk kc landed in LDS
        __syncthreads();                                     // publish; prev buf free
        if (w == 0 && kc < 31) issue_tdm(kk + 32, cur ? lds0 : lds1);

        const f16* as_ = &abuf[cur][0];
        v16h a = *(const v16h*)&as_[arow * 32 + kh2 * 16];
        if (kc < 31) __builtin_prefetch(bbase + kk + 32, 0, 3);  // global_prefetch_b8
#pragma unroll
        for (int i = 0; i < 4; ++i) {
            v16h b = *(const v16h*)(bbase + (size_t)(i * 16) * Z_DIM + kk);
            acc[i] = __builtin_amdgcn_wmma_f32_16x16x32_f16(
                        false, a, false, b, (short)0, acc[i], false, false);
        }
    }

    // Store 4 tiles of 16x16 f32.
#pragma unroll
    for (int i = 0; i < 4; ++i) {
        const int col = n0 + wn * 64 + i * 16 + lane_lo;
#pragma unroll
        for (int r = 0; r < 8; ++r) {
            const int row = m0 + wm * 16 + r + kh2 * 8;
            out[(size_t)row * N_CELLS + col] = acc[i][r];
        }
    }
}

// ---------------------------------------------------------------------------
extern "C" void kernel_launch(void* const* d_in, const int* in_sizes, int n_in,
                              void* d_out, int out_size, void* d_ws, size_t ws_size,
                              hipStream_t stream) {
    (void)in_sizes; (void)n_in; (void)out_size; (void)ws_size;
    const float* Q = (const float*)d_in[0];   // (16384, 128)
    const float* K = (const float*)d_in[1];   // (1024, 128)
    const float* V = (const float*)d_in[2];   // (1024, 4096)

    float* recon  = (float*)d_out;                                   // (16384,4096)
    float* p_attn = (float*)d_out + (size_t)N_GENES * N_CELLS;       // (16384,1024)

    // Workspace layout: Kh (256 KB) | Vt (8 MB) | Ph (32 MB)  => ~42 MB total
    f16* Kh = (f16*)d_ws;
    f16* Vt = (f16*)((char*)d_ws + (size_t)Z_DIM * D_K * sizeof(f16));
    f16* Ph = (f16*)((char*)d_ws + (size_t)Z_DIM * D_K * sizeof(f16)
                                 + (size_t)N_CELLS * Z_DIM * sizeof(f16));

    convert_k_kernel<<<(Z_DIM * D_K) / 256, 256, 0, stream>>>(K, Kh);
    transpose_v_kernel<<<dim3(N_CELLS / 32, Z_DIM / 32), 256, 0, stream>>>(V, Vt);
    scores_softmax_kernel<<<N_GENES / 16, 256, 0, stream>>>(Q, Kh, p_attn, Ph);
    recon_kernel<<<dim3(N_CELLS / 128, N_GENES / 64), 256, 0, stream>>>(Ph, Vt, recon);
}
